// SPQRModule_33251636806448
// MI455X (gfx1250) — matrix operand order repaired and checked by reading, not compiled, not run on hardware
//
#include <hip/hip_runtime.h>

// ---------------------------------------------------------------------------
// SPQR quantized GEMV for MI455X (gfx1250, wave32).
//   y(10,8192) = x(10,8192) @ dequant(W)^T  + CSR outliers
// HBM-bound on one streaming pass over W (256 MB -> ~11us @ 23.3 TB/s).
// Batch dim (10 -> padded 16) = WMMA M; V_WMMA_F32_16X16X32_F16, f32 acc.
// 1024 independent waves (512 row-strips x 2 K-splits) in ONE launch for
// latency hiding; partials combined deterministically with the CSR pass.
// ---------------------------------------------------------------------------

typedef __attribute__((ext_vector_type(16))) _Float16 v16h;
typedef __attribute__((ext_vector_type(8)))  float    v8f;

constexpr int M_DIM  = 8192;          // output rows (W rows)
constexpr int N_DIM  = 8192;          // reduction dim
constexpr int TN     = 512;           // N / 16 tile columns
constexpr int BATCH  = 10;            // real batch rows of x
constexpr int KCH    = 32;            // WMMA K per step
constexpr int NCHUNK = N_DIM / KCH;   // 256 chunks
constexpr int STRIPS = M_DIM / 16;    // 512 row strips (one WMMA N-tile each)

constexpr int    XW_ELEMS  = NCHUNK * 32 * 16;            // packed x fragments
constexpr size_t XW_BYTES  = (size_t)XW_ELEMS * 2;        // 256 KB
constexpr size_t PART_ELEM = (size_t)16 * M_DIM;          // one partial (16 x 8192 f32)
constexpr size_t PART_BYTES = PART_ELEM * 4;              // 512 KB

// ---------------------------------------------------------------------------
// Pack x into WMMA-A fragment-major f16, rows 10..15 zeroed.
// A layout used here (16x32, contiguous k order, same as B): element e of a lane's v16h corresponds to
//   k = kk + 16*half + e,  b = lane&15.
// ---------------------------------------------------------------------------
__global__ __launch_bounds__(256) void pack_x_kernel(
    const float* __restrict__ x, _Float16* __restrict__ xw)
{
  for (int i = blockIdx.x * blockDim.x + threadIdx.x; i < XW_ELEMS;
       i += gridDim.x * blockDim.x) {
    const int e     = i & 15;
    const int lane  = (i >> 4) & 31;
    const int chunk = i >> 9;
    const int half  = lane >> 4;
    const int b     = lane & 15;
    const int k     = chunk * KCH + (half << 4) + e;
    _Float16 v = (_Float16)0.f;
    if (b < BATCH) v = (_Float16)x[(long)b * N_DIM + k];
    xw[i] = v;
  }
}

// ---------------------------------------------------------------------------
// Dense dequant-GEMM. Each wave owns one 16-row strip and one K-split range,
// K-unrolled by 2 so two 32-wide chunks of b128 loads are in flight per body.
// Per chunk: 4 x global_load_b128 (W codes), 1 x 32B A-frag load, two-level
// dequant (pk f32 math -> cvt_pk_f16), one v_wmma_f32_16x16x32_f16.
// ---------------------------------------------------------------------------
template <bool PACKED>
__global__ __launch_bounds__(128) void spqr_wmma_gemv(
    const _Float16* __restrict__ xw,   // packed A fragments (PACKED==true)
    const float*    __restrict__ xf,   // raw x (fallback path)
    const int*      __restrict__ W,    // (8192, 8192) 3-bit codes in int32
    const int*      __restrict__ Ws,   // (8192, 512) quantized scales
    const int*      __restrict__ Wzq,  // (8192, 512) quantized zeros
    const float*    __restrict__ Wss,  // (512, 512) second-order params
    const float*    __restrict__ Wsz,
    const float*    __restrict__ Wzs,
    const float*    __restrict__ Wzz,
    float*          __restrict__ part,      // partial output base
    long                         part_stride, // floats between K-split partials
    int                          nsplit)      // 1 or 2 K-splits
{
  const int lane   = threadIdx.x & 31;
  const int waveId = blockIdx.x * (blockDim.x >> 5) + (threadIdx.x >> 5);
  const int strip  = waveId % STRIPS;       // adjacent waves -> adjacent rows
  const int ks     = waveId / STRIPS;
  if (ks >= nsplit) return;

  const int klen = N_DIM / nsplit;
  const int k0   = ks * klen;
  const int k1   = k0 + klen;
  float* out = part + (long)ks * part_stride;

  const int half = lane >> 4;               // ISA lane-half
  const int l16  = lane & 15;

  // B-fragment ownership: lane holds W row m (== output col n=l16) and a
  // 16-wide consecutive K block (lower half lanes 0..15, upper lanes 16..31).
  const int  m0   = strip * 16;
  const int  m    = m0 + l16;
  const int  tm   = m0 >> 4;                // tile-row index (wave-uniform)
  const long wrow = (long)m * N_DIM;

  v8f c = {};

  for (int kk = k0; kk < k1; kk += 2 * KCH) {
#pragma unroll
    for (int u = 0; u < 2; ++u) {
      const int kc = kk + u * KCH;
      const int kb = kc + (half << 4);      // lane's 16-wide K block
      const int t  = kb >> 4;               // exactly one dequant tile column

      // Two-level dequant parameters for this (row, tile-col).
      const float s = ((float)Ws [(long)m * TN + t] - Wsz[tm * TN + t]) * Wss[tm * TN + t];
      const float z = ((float)Wzq[(long)m * TN + t] - Wzz[tm * TN + t]) * Wzs[tm * TN + t];

      // 16 consecutive codes per lane: 4 x b128.
      const int4* wp = reinterpret_cast<const int4*>(W + wrow + kb);
      v16h bf;
#pragma unroll
      for (int q = 0; q < 4; ++q) {
        const int4 cw = wp[q];
        bf[4 * q + 0] = (_Float16)(((float)cw.x - z) * s);
        bf[4 * q + 1] = (_Float16)(((float)cw.y - z) * s);
        bf[4 * q + 2] = (_Float16)(((float)cw.z - z) * s);
        bf[4 * q + 3] = (_Float16)(((float)cw.w - z) * s);
      }

      // A fragment.
      v16h af;
      if (PACKED) {
        af = *reinterpret_cast<const v16h*>(
            xw + (((long)(kc >> 5) * 32 + lane) << 4));   // one contiguous 32B
      } else {
#pragma unroll
        for (int e = 0; e < 16; ++e) {
          const int k = kc + (half << 4) + e;
          af[e] = (l16 < BATCH) ? (_Float16)xf[(long)l16 * N_DIM + k]
                                : (_Float16)0.f;
        }
      }

      // Pull the next body's chunk toward L2/WGP$ (global_prefetch_b8,
      // no LOADcnt pressure).
      __builtin_prefetch(W + wrow + kb + 2 * KCH, 0, 1);

      c = __builtin_amdgcn_wmma_f32_16x16x32_f16(false, af, false, bf,
                                                 (short)0, c, false, false);
    }
  }

  // C layout: VGPR r, lanes 0..15 -> M=r (batch), lanes 16..31 -> M=8+r.
#pragma unroll
  for (int r = 0; r < 8; ++r) {
    const int bb = (half << 3) + r;
    if (bb < BATCH) out[(long)bb * M_DIM + m] = c[r];
  }
}

// ---------------------------------------------------------------------------
// CSR outliers + deterministic combine:  y = p0 (+ p1) + outlier correction.
// One wave32 per output row; shfl_xor tree reduction; lane 0 sole writer.
// Fallback path passes p0 == y, has_p1 == 0 (in-place accumulate).
// ---------------------------------------------------------------------------
__global__ __launch_bounds__(128) void spqr_csr_combine_kernel(
    const float* __restrict__ x,
    const int*   __restrict__ row_offsets,
    const int*   __restrict__ col_ids,
    const float* __restrict__ vals,
    const float* __restrict__ p0,
    const float* __restrict__ p1,
    int                       has_p1,
    float*       __restrict__ y)
{
  const int lane = threadIdx.x & 31;
  const int m    = blockIdx.x * (blockDim.x >> 5) + (threadIdx.x >> 5);
  if (m >= M_DIM) return;

  const int r0 = row_offsets[m];
  const int r1 = row_offsets[m + 1];

  // Common case: exactly 32 nnz -> one element per lane, held in registers.
  int   col = 0;
  float v   = 0.f;
  const int j = r0 + lane;
  if (j < r1) { col = col_ids[j]; v = vals[j]; }

#pragma unroll
  for (int bb = 0; bb < BATCH; ++bb) {
    float acc = v * x[(long)bb * N_DIM + col];
    for (int jj = r0 + lane + 32; jj < r1; jj += 32)   // generic tail
      acc += vals[jj] * x[(long)bb * N_DIM + col_ids[jj]];
#pragma unroll
    for (int off = 16; off > 0; off >>= 1)
      acc += __shfl_xor(acc, off, 32);
    if (lane == 0) {
      const long idx = (long)bb * M_DIM + m;
      float sum = acc + p0[idx];
      if (has_p1) sum += p1[idx];
      y[idx] = sum;
    }
  }
}

// ---------------------------------------------------------------------------
extern "C" void kernel_launch(void* const* d_in, const int* in_sizes, int n_in,
                              void* d_out, int out_size, void* d_ws,
                              size_t ws_size, hipStream_t stream)
{
  (void)in_sizes; (void)n_in; (void)out_size;

  const float* x   = (const float*)d_in[0];
  const int*   W   = (const int*)  d_in[1];
  const int*   Ws  = (const int*)  d_in[2];
  const int*   Wzq = (const int*)  d_in[3];
  const float* Wss = (const float*)d_in[4];
  const float* Wsz = (const float*)d_in[5];
  const float* Wzs = (const float*)d_in[6];
  const float* Wzz = (const float*)d_in[7];
  const int*   ro  = (const int*)  d_in[8];
  const int*   ci  = (const int*)  d_in[9];
  const float* vv  = (const float*)d_in[10];
  float*       y   = (float*)d_out;

  const dim3 blk(128);                       // 4 wave32 per block

  if (ws_size >= XW_BYTES + 2 * PART_BYTES) {
    // Full path: packed x, 2-way K split -> 1024 waves in one launch.
    _Float16* xw = (_Float16*)d_ws;
    float*    p0 = (float*)((char*)d_ws + XW_BYTES);
    float*    p1 = (float*)((char*)d_ws + XW_BYTES + PART_BYTES);
    pack_x_kernel<<<dim3(128), dim3(256), 0, stream>>>(x, xw);
    spqr_wmma_gemv<true><<<dim3(STRIPS * 2 / 4), blk, 0, stream>>>(
        xw, x, W, Ws, Wzq, Wss, Wsz, Wzs, Wzz,
        p0, (long)PART_ELEM, /*nsplit=*/2);
    spqr_csr_combine_kernel<<<dim3(M_DIM / 4), blk, 0, stream>>>(
        x, ro, ci, vv, p0, p1, /*has_p1=*/1, y);
  } else if (ws_size >= XW_BYTES) {
    // Packed x, single K pass straight into y.
    _Float16* xw = (_Float16*)d_ws;
    pack_x_kernel<<<dim3(128), dim3(256), 0, stream>>>(x, xw);
    spqr_wmma_gemv<true><<<dim3(STRIPS / 4), blk, 0, stream>>>(
        xw, x, W, Ws, Wzq, Wss, Wsz, Wzs, Wzz, y, 0L, /*nsplit=*/1);
    spqr_csr_combine_kernel<<<dim3(M_DIM / 4), blk, 0, stream>>>(
        x, ro, ci, vv, y, y, /*has_p1=*/0, y);
  } else {
    // Minimal-scratch fallback: inline f32 x loads.
    spqr_wmma_gemv<false><<<dim3(STRIPS / 4), blk, 0, stream>>>(
        nullptr, x, W, Ws, Wzq, Wss, Wsz, Wzs, Wzz, y, 0L, /*nsplit=*/1);
    spqr_csr_combine_kernel<<<dim3(M_DIM / 4), blk, 0, stream>>>(
        x, ro, ci, vv, y, y, /*has_p1=*/0, y);
  }
}
